// Mamba_block_29008209117720
// MI455X (gfx1250) — matrix operand order repaired and checked by reading, not compiled, not run on hardware
//
#include <hip/hip_runtime.h>
#include <hip/hip_bf16.h>

// ---------------------------------------------------------------------------
// Mamba block for MI455X (gfx1250, wave32).
// GEMMs in bf16 via v_wmma_f32_16x16x32_bf16 (f32 accumulate), 2x2 register
// blocking per wave, WGP-scope prefetch on the streaming rows; RMSNorm /
// causal conv / softplus / selective scan / SiLU gates in fp32.
// ---------------------------------------------------------------------------

#define BB 2
#define LL 1024
#define DM 1024
#define DS 16
#define DC 4
#define DI 2048          // E * DM
#define DR 64            // DM / 16
#define NROW (BB * LL)   // 2048 token rows
#define EPS 1e-5f

typedef __attribute__((ext_vector_type(16))) __bf16 v16bf;
typedef __attribute__((ext_vector_type(8)))  float  v8f;

union FragAB {
    uint4 u4[2];
    v16bf v;
};

__device__ __forceinline__ unsigned short f2bf(float f) {
    unsigned int u = __float_as_uint(f);
    unsigned int r = u + 0x7FFFu + ((u >> 16) & 1u);   // round-to-nearest-even
    return (unsigned short)(r >> 16);
}

__device__ __forceinline__ float softplusf(float x) {
    return fmaxf(x, 0.0f) + log1pf(__expf(-fabsf(x)));
}

__device__ __forceinline__ float siluf(float x) {
    return x / (1.0f + __expf(-x));
}

// ---------------------------------------------------------------------------
// 1) RMSNorm: x[NROW, DM] f32 -> u bf16
// ---------------------------------------------------------------------------
__global__ void k_rmsnorm_bf16(const float* __restrict__ x,
                               const float* __restrict__ norm_w,
                               unsigned short* __restrict__ u_bf) {
    const int row = blockIdx.x;
    const int tid = threadIdx.x;           // 256 threads, 4 elems each
    const float* xr = x + (size_t)row * DM;

    float ss = 0.f;
    float vals[4];
#pragma unroll
    for (int k = 0; k < 4; ++k) {
        float v = xr[tid + k * 256];
        vals[k] = v;
        ss += v * v;
    }
    __shared__ float red[256];
    red[tid] = ss;
    __syncthreads();
    for (int s = 128; s > 0; s >>= 1) {
        if (tid < s) red[tid] += red[tid + s];
        __syncthreads();
    }
    const float scale = rsqrtf(red[0] * (1.0f / DM) + EPS);
#pragma unroll
    for (int k = 0; k < 4; ++k) {
        int c = tid + k * 256;
        u_bf[(size_t)row * DM + c] = f2bf(vals[k] * scale * norm_w[c]);
    }
}

// ---------------------------------------------------------------------------
// 2) fp32 -> bf16 elementwise convert (weights, dt slice)
// ---------------------------------------------------------------------------
__global__ void k_f2bf(const float* __restrict__ in,
                       unsigned short* __restrict__ out, int n) {
    int i = blockIdx.x * blockDim.x + threadIdx.x;
    if (i < n) out[i] = f2bf(in[i]);
}

// slice dt = dbl[:, 0:DR] -> bf16 [NROW, DR]
__global__ void k_dt_slice_bf16(const float* __restrict__ dbl,
                                unsigned short* __restrict__ dt_bf) {
    int i = blockIdx.x * blockDim.x + threadIdx.x;  // NROW*DR
    if (i >= NROW * DR) return;
    int row = i / DR, c = i % DR;
    dt_bf[i] = f2bf(dbl[(size_t)row * 96 + c]);
}

// ---------------------------------------------------------------------------
// Fragment loaders for C = A * W^T  (A,W row-major bf16)
//   16-bit A 16x32 layout: lanes 0-15 hold K = k+0..7 / k+16..23,
//                          lanes 16-31 hold K = k+8..15 / k+24..31.
//   16-bit B 32x16 layout: lane = column, same K split as A.
// ---------------------------------------------------------------------------
__device__ __forceinline__ v16bf load_fragA(const unsigned short* p) {
    FragAB f;
    f.u4[0] = *(const uint4*)(p);
    f.u4[1] = *(const uint4*)(p + 16);
    return f.v;
}
__device__ __forceinline__ v16bf load_fragB(const unsigned short* p) {
    FragAB f;
    f.u4[0] = *(const uint4*)(p);
    f.u4[1] = *(const uint4*)(p + 16);
    return f.v;
}

__device__ __forceinline__ float gemm_epilogue(float v, int mode, int row, int col,
                                               int N, const float* bias,
                                               const float* resid) {
    if (mode == 1)      v = softplusf(v + bias[col]);
    else if (mode == 2) v += resid[(size_t)row * N + col];
    return v;
}

// ---------------------------------------------------------------------------
// 3a) bf16 GEMM, 2x2 tiles per wave (32x32), 8 waves -> 64x128 block tile.
//     Requires M % 64 == 0, N % 128 == 0, K % 32 == 0.
//     mode 0: C = acc ; mode 1: softplus(acc+bias[col]) ; mode 2: acc+resid.
// ---------------------------------------------------------------------------
__global__ void k_gemm_bf16_2x2(const unsigned short* __restrict__ A,
                                const unsigned short* __restrict__ W,
                                float* __restrict__ C,
                                int M, int N, int K, int mode,
                                const float* __restrict__ bias,
                                const float* __restrict__ resid) {
    const int lane = threadIdx.x & 31;
    const int wave = threadIdx.x >> 5;
    const int wm = wave >> 2;               // 0..1
    const int wn = wave & 3;                // 0..3
    const int m0 = blockIdx.y * 64 + wm * 32;
    const int n0 = blockIdx.x * 128 + wn * 32;

    const int half = lane >> 4;
    const int lr   = lane & 15;

    const unsigned short* arow0 = A + (size_t)(m0 + lr) * K + half * 8;
    const unsigned short* arow1 = arow0 + (size_t)16 * K;
    const unsigned short* wrow0 = W + (size_t)(n0 + lr) * K + half * 8;
    const unsigned short* wrow1 = wrow0 + (size_t)16 * K;

    v8f acc00 = {}, acc01 = {}, acc10 = {}, acc11 = {};

    for (int k0 = 0; k0 < K; k0 += 32) {
        // WGP-scope prefetch 4 K-steps (128 elements / 256B) ahead, once per
        // 128B cacheline (the &63 test folds statically in the unrolled body).
        if ((k0 & 63) == 0) {
            __builtin_prefetch(arow0 + k0 + 128, 0, 3);
            __builtin_prefetch(arow1 + k0 + 128, 0, 3);
            __builtin_prefetch(wrow0 + k0 + 128, 0, 3);
            __builtin_prefetch(wrow1 + k0 + 128, 0, 3);
        }

        v16bf a0 = load_fragA(arow0 + k0);
        v16bf a1 = load_fragA(arow1 + k0);
        v16bf b0 = load_fragB(wrow0 + k0);
        v16bf b1 = load_fragB(wrow1 + k0);

        acc00 = __builtin_amdgcn_wmma_f32_16x16x32_bf16(false, a0, false, b0,
                                                        (short)0, acc00, false, false);
        acc01 = __builtin_amdgcn_wmma_f32_16x16x32_bf16(false, a0, false, b1,
                                                        (short)0, acc01, false, false);
        acc10 = __builtin_amdgcn_wmma_f32_16x16x32_bf16(false, a1, false, b0,
                                                        (short)0, acc10, false, false);
        acc11 = __builtin_amdgcn_wmma_f32_16x16x32_bf16(false, a1, false, b1,
                                                        (short)0, acc11, false, false);
    }

    // C layout per tile: VGPR i -> row tile_m + i + 8*half, col tile_n + lr
#pragma unroll
    for (int i = 0; i < 8; ++i) {
        const int r0 = m0 + i + half * 8;
        const int r1 = r0 + 16;
        const int c0 = n0 + lr;
        const int c1 = c0 + 16;
        C[(size_t)r0 * N + c0] = gemm_epilogue(acc00[i], mode, r0, c0, N, bias, resid);
        C[(size_t)r0 * N + c1] = gemm_epilogue(acc01[i], mode, r0, c1, N, bias, resid);
        C[(size_t)r1 * N + c0] = gemm_epilogue(acc10[i], mode, r1, c0, N, bias, resid);
        C[(size_t)r1 * N + c1] = gemm_epilogue(acc11[i], mode, r1, c1, N, bias, resid);
    }
}

// ---------------------------------------------------------------------------
// 3b) bf16 GEMM, one 16x16 tile per wave (for small N like x_proj's N=96)
// ---------------------------------------------------------------------------
__global__ void k_gemm_bf16_small(const unsigned short* __restrict__ A,
                                  const unsigned short* __restrict__ W,
                                  float* __restrict__ C,
                                  int M, int N, int K) {
    const int lane = threadIdx.x & 31;
    const int wave = threadIdx.x >> 5;
    const int m0 = blockIdx.y * 16;
    const int n0 = blockIdx.x * 128 + wave * 16;
    if (n0 >= N) return;                    // uniform across the wave

    const int half = lane >> 4;
    const int lr   = lane & 15;

    const unsigned short* arow = A + (size_t)(m0 + lr) * K + half * 8;
    const unsigned short* wrow = W + (size_t)(n0 + lr) * K + half * 8;

    v8f acc = {};
    for (int k0 = 0; k0 < K; k0 += 32) {
        if ((k0 & 63) == 0) {
            __builtin_prefetch(arow + k0 + 128, 0, 3);
            __builtin_prefetch(wrow + k0 + 128, 0, 3);
        }
        v16bf a = load_fragA(arow + k0);
        v16bf b = load_fragB(wrow + k0);
        acc = __builtin_amdgcn_wmma_f32_16x16x32_bf16(false, a, false, b,
                                                      (short)0, acc, false, false);
    }
    const int col = n0 + lr;
#pragma unroll
    for (int i = 0; i < 8; ++i) {
        int row = m0 + i + half * 8;
        C[(size_t)row * N + col] = acc[i];
    }
}

// ---------------------------------------------------------------------------
// 4) causal depthwise conv (DC=4) + bias + SiLU.
//    xz[NROW, 2*DI]: cols [0,DI) = xin branch. Writes xc f32 and xc bf16.
// ---------------------------------------------------------------------------
__global__ void k_conv_silu(const float* __restrict__ xz,
                            const float* __restrict__ conv_w,
                            const float* __restrict__ conv_b,
                            float* __restrict__ xc,
                            unsigned short* __restrict__ xc_bf) {
    int i = blockIdx.x * blockDim.x + threadIdx.x;   // B*L*DI
    if (i >= NROW * DI) return;
    int d = i % DI;
    int r = i / DI;            // b*L + l
    int l = r % LL;
    int rbase = r - l;         // b*L

    float acc = conv_b[d];
#pragma unroll
    for (int w = 0; w < DC; ++w) {
        int lt = l - (DC - 1) + w;
        float xv = (lt >= 0) ? xz[(size_t)(rbase + lt) * (2 * DI) + d] : 0.0f;
        acc += xv * conv_w[d * DC + w];
    }
    float v = siluf(acc);
    xc[(size_t)r * DI + d] = v;
    xc_bf[(size_t)r * DI + d] = f2bf(v);
}

// ---------------------------------------------------------------------------
// 5) selective scan fused with output gate.
//    One thread per (b, channel d); h[16] in registers; B/C staged in LDS
//    per 8-timestep chunk. Writes y_gated bf16 = (y + xc*D) * silu(z).
// ---------------------------------------------------------------------------
__global__ void k_scan(const float* __restrict__ delta,
                       const float* __restrict__ dbl,
                       const float* __restrict__ xz,
                       const float* __restrict__ xc,
                       const float* __restrict__ A_log,
                       const float* __restrict__ Dp,
                       unsigned short* __restrict__ y_bf) {
    const int tid = threadIdx.x;                 // 256
    const int d = blockIdx.x * 256 + tid;        // channel
    const int b = blockIdx.y;

    float a[DS], h[DS];
#pragma unroll
    for (int n = 0; n < DS; ++n) {
        a[n] = -__expf(A_log[(size_t)d * DS + n]);
        h[n] = 0.0f;
    }
    const float Dd = Dp[d];

    __shared__ float sB[8][DS];
    __shared__ float sC[8][DS];

    for (int t0 = 0; t0 < LL; t0 += 8) {
        // stage B_t, C_t for 8 timesteps: 8*32 floats, one per thread
        {
            int tt = tid >> 5;
            int e  = tid & 31;
            float v = dbl[(size_t)(b * LL + t0 + tt) * 96 + DR + e];
            if (e < DS) sB[tt][e] = v;
            else        sC[tt][e - DS] = v;
        }
        __syncthreads();

#pragma unroll 1
        for (int j = 0; j < 8; ++j) {
            const size_t r = (size_t)(b * LL + t0 + j);
            const float dlt = delta[r * DI + d];
            const float xv  = xc[r * DI + d];
            float y = 0.0f;
#pragma unroll
            for (int n = 0; n < DS; ++n) {
                float dA = __expf(dlt * a[n]);
                h[n] = dA * h[n] + dlt * sB[j][n] * xv;
                y += h[n] * sC[j][n];
            }
            y += xv * Dd;
            const float zv = xz[r * (2 * DI) + DI + d];
            y_bf[r * DI + d] = f2bf(y * siluf(zv));
        }
        __syncthreads();
    }
}

// ---------------------------------------------------------------------------
// launch
// ---------------------------------------------------------------------------
extern "C" void kernel_launch(void* const* d_in, const int* in_sizes, int n_in,
                              void* d_out, int out_size, void* d_ws, size_t ws_size,
                              hipStream_t stream) {
    const float* x        = (const float*)d_in[0];   // [B,L,DM]
    const float* norm_w   = (const float*)d_in[1];   // [DM]
    const float* in_w     = (const float*)d_in[2];   // [2*DI, DM]
    const float* conv_w   = (const float*)d_in[3];   // [DI,1,DC]
    const float* conv_b   = (const float*)d_in[4];   // [DI]
    const float* xp_w     = (const float*)d_in[5];   // [96, DI]
    const float* dt_w     = (const float*)d_in[6];   // [DI, DR]
    const float* dt_b     = (const float*)d_in[7];   // [DI]
    const float* A_log    = (const float*)d_in[8];   // [DI, DS]
    const float* Dp       = (const float*)d_in[9];   // [DI]
    const float* out_w    = (const float*)d_in[10];  // [DM, DI]
    float* out            = (float*)d_out;           // [B,L,DM]

    // ---- workspace bump allocator (256B aligned) ----
    char* ws = (char*)d_ws;
    size_t off = 0;
    auto alloc = [&](size_t bytes) -> void* {
        off = (off + 255) & ~(size_t)255;
        void* p = ws + off;
        off += bytes;
        return p;
    };
    unsigned short* u_bf    = (unsigned short*)alloc((size_t)NROW * DM * 2);
    unsigned short* w_in_bf = (unsigned short*)alloc((size_t)2 * DI * DM * 2);
    unsigned short* w_xp_bf = (unsigned short*)alloc((size_t)96 * DI * 2);
    unsigned short* w_dt_bf = (unsigned short*)alloc((size_t)DI * DR * 2);
    unsigned short* w_out_bf= (unsigned short*)alloc((size_t)DM * DI * 2);
    float*          xz      = (float*)alloc((size_t)NROW * 2 * DI * 4);
    float*          xc      = (float*)alloc((size_t)NROW * DI * 4);
    unsigned short* xc_bf   = (unsigned short*)alloc((size_t)NROW * DI * 2);
    float*          dbl     = (float*)alloc((size_t)NROW * 96 * 4);
    unsigned short* dt_bf   = (unsigned short*)alloc((size_t)NROW * DR * 2);
    float*          delta   = (float*)alloc((size_t)NROW * DI * 4);
    unsigned short* y_bf    = (unsigned short*)alloc((size_t)NROW * DI * 2);
    (void)ws_size; (void)in_sizes; (void)n_in; (void)out_size;

    // 1) RMSNorm -> bf16 activations
    k_rmsnorm_bf16<<<NROW, 256, 0, stream>>>(x, norm_w, u_bf);

    // 2) weight converts to bf16
    auto cvt = [&](const float* src, unsigned short* dst, int n) {
        k_f2bf<<<(n + 255) / 256, 256, 0, stream>>>(src, dst, n);
    };
    cvt(in_w,  w_in_bf,  2 * DI * DM);
    cvt(xp_w,  w_xp_bf,  96 * DI);
    cvt(dt_w,  w_dt_bf,  DI * DR);
    cvt(out_w, w_out_bf, DM * DI);

    // 3) in_proj: xz[2048,4096] = u * in_w^T
    k_gemm_bf16_2x2<<<dim3((2 * DI) / 128, NROW / 64), 256, 0, stream>>>(
        u_bf, w_in_bf, xz, NROW, 2 * DI, DM, 0, nullptr, nullptr);

    // 4) causal conv + SiLU
    k_conv_silu<<<(NROW * DI) / 256, 256, 0, stream>>>(xz, conv_w, conv_b, xc, xc_bf);

    // 5) x_proj: dbl[2048,96] = xc * xp_w^T   (N=96 -> small kernel)
    k_gemm_bf16_small<<<dim3(1, NROW / 16), 256, 0, stream>>>(
        xc_bf, w_xp_bf, dbl, NROW, 96, DI);

    // 6) dt slice -> bf16
    k_dt_slice_bf16<<<(NROW * DR + 255) / 256, 256, 0, stream>>>(dbl, dt_bf);

    // 7) dt_proj + bias + softplus: delta[2048,2048]
    k_gemm_bf16_2x2<<<dim3(DI / 128, NROW / 64), 256, 0, stream>>>(
        dt_bf, w_dt_bf, delta, NROW, DI, DR, 1, dt_b, nullptr);

    // 8) selective scan + D skip + SiLU gate -> y_bf
    k_scan<<<dim3(DI / 256, BB), 256, 0, stream>>>(
        delta, dbl, xz, xc, A_log, Dp, y_bf);

    // 9) out_proj + residual: out[2048,1024]
    k_gemm_bf16_2x2<<<dim3(DM / 128, NROW / 64), 256, 0, stream>>>(
        y_bf, w_out_bf, out, NROW, DM, DI, 2, nullptr, x);
}
